// DecoderLM_51213190038036
// MI455X (gfx1250) — matrix operand order repaired and checked by reading, not compiled, not run on hardware
//
#include <hip/hip_runtime.h>

// ---------------- problem constants (GPT-2 small fwd, B=2, S=1024) ----------
constexpr int Vv    = 50257;
constexpr int Dd    = 768;
constexpr int Hh    = 12;
constexpr int Ll    = 12;
constexpr int Bb    = 2;
constexpr int Ss    = 1024;
constexpr int Mrows = Bb * Ss;     // 2048 token rows
constexpr int DFF   = 4 * Dd;      // 3072

// ---------------- types -----------------------------------------------------
typedef __attribute__((ext_vector_type(16))) __bf16       v16bf;
typedef __attribute__((ext_vector_type(8)))  float        v8f;
typedef __attribute__((ext_vector_type(4)))  unsigned int u32x4;
typedef __attribute__((ext_vector_type(4)))  float        f32x4;

struct Frag256 { u32x4 lo, hi; };   // 32B, trivially copyable -> bit_cast v16bf

// fp32 -> bf16 via native convert (backend packs v_cvt_pk_bf16_f32)
__device__ __forceinline__ unsigned short f2bf(float f) {
  return __builtin_bit_cast(unsigned short, static_cast<__bf16>(f));
}

// two 16B-aligned 8xbf16 chunks -> one v16bf fragment
__device__ __forceinline__ v16bf load_frag16(const unsigned short* p_lo,
                                             const unsigned short* p_hi) {
  Frag256 f;
  f.lo = *(const u32x4*)p_lo;
  f.hi = *(const u32x4*)p_hi;
  return __builtin_bit_cast(v16bf, f);
}

// 16 bf16 gathered with an element stride (for V^T fragments from LDS)
__device__ __forceinline__ v16bf load_frag_strided(const unsigned short* p, int stride) {
  Frag256 f;
#pragma unroll
  for (int j = 0; j < 4; ++j)
    f.lo[j] = (unsigned)p[(2 * j) * stride] | ((unsigned)p[(2 * j + 1) * stride] << 16);
#pragma unroll
  for (int j = 0; j < 4; ++j)
    f.hi[j] = (unsigned)p[(8 + 2 * j) * stride] | ((unsigned)p[(8 + 2 * j + 1) * stride] << 16);
  return __builtin_bit_cast(v16bf, f);
}

// CDNA5 async global->LDS copy (ASYNCcnt tracked); LDS offset = low 32 bits of
// the generic pointer (flat LDS aperture keeps the wave-relative offset there).
__device__ __forceinline__ void async_copy_b128(const unsigned short* gsrc,
                                                unsigned short* lds_dst) {
  const unsigned loff = (unsigned)(unsigned long long)(uintptr_t)lds_dst;
  const unsigned long long gaddr = (unsigned long long)(uintptr_t)gsrc;
  asm volatile("global_load_async_to_lds_b128 %0, %1, off"
               :: "v"(loff), "v"(gaddr)
               : "memory");
}

__device__ __forceinline__ void wait_async0() {
  asm volatile("s_wait_asynccnt 0" ::: "memory");
}
__device__ __forceinline__ void wait_ds0() {
  asm volatile("s_wait_dscnt 0" ::: "memory");
}

__device__ __forceinline__ float gelu_exact(float x) {
  return 0.5f * x * (1.0f + erff(x * 0.70710678118654752f));
}

// ---------------- position-id scan (cumsum of mask - 1, masked -> 0) ---------
__global__ __launch_bounds__(1024) void pos_scan_kernel(
    const float* __restrict__ amask, int* __restrict__ pos_ids) {
  __shared__ float sm[Ss];
  const int b = blockIdx.x;
  const int t = threadIdx.x;
  const float v = amask[b * Ss + t];
  sm[t] = v;
  __syncthreads();
  for (int off = 1; off < Ss; off <<= 1) {
    float add = (t >= off) ? sm[t - off] : 0.0f;
    __syncthreads();
    sm[t] += add;
    __syncthreads();
  }
  float p = (v == 0.0f) ? 0.0f : (sm[t] - 1.0f);
  pos_ids[b * Ss + t] = (int)p;
}

// ---------------- embedding gather ------------------------------------------
__global__ __launch_bounds__(256) void embed_kernel(
    const int* __restrict__ ids, const int* __restrict__ pos_ids,
    const float* __restrict__ tok, const float* __restrict__ pos,
    float* __restrict__ x) {
  const int i  = blockIdx.x * blockDim.x + threadIdx.x;
  const int t  = i / (Dd / 4);
  const int dv = i % (Dd / 4);
  if (t >= Mrows) return;
  const f32x4 te = ((const f32x4*)(tok + (size_t)ids[t] * Dd))[dv];
  const f32x4 pe = ((const f32x4*)(pos + (size_t)pos_ids[t] * Dd))[dv];
  ((f32x4*)(x + (size_t)t * Dd))[dv] = te + pe;
}

// ---------------- LayerNorm: fp32 in -> bf16 out -----------------------------
__global__ __launch_bounds__(256) void ln_kernel(
    const float* __restrict__ x, const float* __restrict__ g,
    const float* __restrict__ bta, unsigned short* __restrict__ y) {
  __shared__ float red[256];
  const int r = blockIdx.x;
  const int t = threadIdx.x;
  const float* xr = x + (size_t)r * Dd;
  float s = 0.0f;
  for (int i = t; i < Dd; i += 256) s += xr[i];
  red[t] = s; __syncthreads();
  for (int off = 128; off >= 1; off >>= 1) { if (t < off) red[t] += red[t + off]; __syncthreads(); }
  const float mu = red[0] / (float)Dd;
  __syncthreads();
  float v = 0.0f;
  for (int i = t; i < Dd; i += 256) { float d = xr[i] - mu; v += d * d; }
  red[t] = v; __syncthreads();
  for (int off = 128; off >= 1; off >>= 1) { if (t < off) red[t] += red[t + off]; __syncthreads(); }
  const float rstd = rsqrtf(red[0] / (float)Dd + 1e-5f);
  for (int i = t; i < Dd; i += 256)
    y[(size_t)r * Dd + i] = f2bf((xr[i] - mu) * rstd * g[i] + bta[i]);
}

// ---------------- generic WMMA GEMM -----------------------------------------
// C[M,N] = A(bf16,[M,K]) @ W(fp32) + bias ; optional GELU ; out fp32(+residual)
// or bf16. wlayout: 0 -> W is (K,N) row-major ; 1 -> W is (N,K) (tied LM head).
// Grid: x = M tiles (fast -> consecutive blocks reuse the same W tile in L2),
//       y = N tiles.
constexpr int BM = 64, BN = 64, BK = 64;
constexpr int LDT = 72;   // ushort stride: rows are 144B (16B-aligned chunks)

__global__ __launch_bounds__(256) void gemm_bf16_kernel(
    const unsigned short* __restrict__ A,
    const float* __restrict__ W,
    const float* __restrict__ bias,
    const float* __restrict__ residual,
    float* __restrict__ outF,
    unsigned short* __restrict__ outB,
    int M, int N, int K, int wlayout, int act) {
  __shared__ unsigned short lA[BM * LDT];
  __shared__ unsigned short lB[BN * LDT];   // transposed: [n][k]

  const int tid  = threadIdx.x;
  const int lane = tid & 31;
  const int wid  = tid >> 5;     // 8 waves
  const int wm   = wid & 3;      // 4 M-tiles of 16
  const int wn   = wid >> 2;     // 2 N-chunks of 32
  const int bm   = blockIdx.x * BM;
  const int bn   = blockIdx.y * BN;
  const int row  = lane & 15;
  const int half = lane >> 4;

  // staging decompositions (256 threads)
  const int ar = tid >> 2;             // A: row 0..63
  const int ac = (tid & 3) * 16;       // A: k offset {0,16,32,48}
  const int bkk = tid >> 2;            // wlayout0: k row 0..63
  const int bn0 = (tid & 3) * 16;      // wlayout0: n offset {0,16,32,48}
  const int tn  = tid >> 2;            // wlayout1: n row 0..63
  const int tks = (tid & 3) * 16;      // wlayout1: k offset

  v8f acc0 = {}, acc1 = {};

  for (int k0 = 0; k0 < K; k0 += BK) {
    // ---- A tile 64x64 bf16: two async b128 copies per thread ----
    {
      const unsigned short* gsrc = A + (size_t)(bm + ar) * K + k0 + ac;
      unsigned short* ldst = &lA[ar * LDT + ac];
      async_copy_b128(gsrc, ldst);
      async_copy_b128(gsrc + 8, ldst + 8);
    }
    // ---- W tile 64x64 fp32 -> bf16, stored transposed lB[n][k] ----
    if (wlayout == 0) {
      const float* src = W + (size_t)(k0 + bkk) * N + bn + bn0;
      if (k0 + BK < K) __builtin_prefetch(src + (size_t)BK * N, 0, 1);
#pragma unroll
      for (int j = 0; j < 16; ++j) {
        const int n = bn + bn0 + j;
        lB[(bn0 + j) * LDT + bkk] = f2bf((n < N) ? src[j] : 0.0f);
      }
    } else {
      const int n = bn + tn;
      if (n < N) {
        const float* src = W + (size_t)n * K + k0 + tks;
        if (k0 + BK < K) __builtin_prefetch(src + BK, 0, 1);
#pragma unroll
        for (int j = 0; j < 16; ++j) lB[tn * LDT + tks + j] = f2bf(src[j]);
      } else {
#pragma unroll
        for (int j = 0; j < 16; ++j) lB[tn * LDT + tks + j] = 0;
      }
    }
    wait_async0();
    __syncthreads();

    // fragments in contiguous k order (k = 16*half + s, same for A and B); two K=32 steps per stage
    const unsigned short* ab = &lA[(wm * 16 + row) * LDT];
    const v16bf af0 = load_frag16(ab + half * 16,      ab + half * 16 + 8);
    const v16bf af1 = load_frag16(ab + 32 + half * 16, ab + 40 + half * 16);
    const unsigned short* b0 = &lB[(wn * 32 + row) * LDT];
    const unsigned short* b1 = &lB[(wn * 32 + 16 + row) * LDT];
    const v16bf b0c0 = load_frag16(b0 + half * 16,      b0 + half * 16 + 8);
    const v16bf b0c1 = load_frag16(b0 + 32 + half * 16, b0 + 40 + half * 16);
    const v16bf b1c0 = load_frag16(b1 + half * 16,      b1 + half * 16 + 8);
    const v16bf b1c1 = load_frag16(b1 + 32 + half * 16, b1 + 40 + half * 16);

    acc0 = __builtin_amdgcn_wmma_f32_16x16x32_bf16(false, af0, false, b0c0, (short)0, acc0, false, false);
    acc0 = __builtin_amdgcn_wmma_f32_16x16x32_bf16(false, af1, false, b0c1, (short)0, acc0, false, false);
    acc1 = __builtin_amdgcn_wmma_f32_16x16x32_bf16(false, af0, false, b1c0, (short)0, acc1, false, false);
    acc1 = __builtin_amdgcn_wmma_f32_16x16x32_bf16(false, af1, false, b1c1, (short)0, acc1, false, false);
    __syncthreads();
  }

  const int col0 = bn + wn * 32 + row;
  const int col1 = col0 + 16;
  const float bias0 = (bias && col0 < N) ? bias[col0] : 0.0f;
  const float bias1 = (bias && col1 < N) ? bias[col1] : 0.0f;
#pragma unroll
  for (int r = 0; r < 8; ++r) {
    const int gr = bm + wm * 16 + 8 * half + r;
    float v0 = acc0[r] + bias0;
    float v1 = acc1[r] + bias1;
    if (act == 1) { v0 = gelu_exact(v0); v1 = gelu_exact(v1); }
    if (outB) {
      if (col0 < N) outB[(size_t)gr * N + col0] = f2bf(v0);
      if (col1 < N) outB[(size_t)gr * N + col1] = f2bf(v1);
    } else {
      if (col0 < N) {
        if (residual) v0 += residual[(size_t)gr * N + col0];
        outF[(size_t)gr * N + col0] = v0;
      }
      if (col1 < N) {
        if (residual) v1 += residual[(size_t)gr * N + col1];
        outF[(size_t)gr * N + col1] = v1;
      }
    }
  }
}

// ---------------- flash attention (causal + key mask), bf16 WMMA -------------
// one wave per 16-query block per (b,h); 32 keys per iteration.
__global__ __launch_bounds__(128) void flash_attn_kernel(
    const unsigned short* __restrict__ Q,
    const unsigned short* __restrict__ Kmat,
    const unsigned short* __restrict__ Vmat,
    const float* __restrict__ amask,
    unsigned short* __restrict__ O) {
  __shared__ unsigned short Vlds[4][32 * 72];  // [key][dim], 144B rows
  __shared__ unsigned short Plds[4][16 * 40];  // probs tile, C->A re-lane

  const int lane = threadIdx.x & 31;
  const int wid  = threadIdx.x >> 5;
  const int w    = blockIdx.x * 4 + wid;
  const int b    = w / (Hh * (Ss / 16));
  const int rem  = w % (Hh * (Ss / 16));
  const int h    = rem / (Ss / 16);
  const int qb   = rem % (Ss / 16);
  const int qbase = qb * 16;

  const int row  = lane & 15;
  const int half = lane >> 4;

  // Q as two A-fragments (HD=64 split into K=32 chunks)
  const unsigned short* qp = Q + (size_t)(b * Ss + qbase + row) * Dd + h * 64;
  const v16bf qf0 = load_frag16(qp + half * 16,      qp + half * 16 + 8);
  const v16bf qf1 = load_frag16(qp + 32 + half * 16, qp + 40 + half * 16);

  float m[8], lsum[8];
  v8f o[4] = {};
#pragma unroll
  for (int r = 0; r < 8; ++r) { m[r] = -3.0e38f; lsum[r] = 0.0f; }

  const float scale = 0.125f;        // 1/sqrt(64)
  const int kend = qbase + 16;       // causal upper bound (exclusive)

  for (int kbase = 0; kbase < kend; kbase += 32) {
    {  // stage V tile (32 keys x 64 dims), one key row per lane, async b128
      const unsigned short* vsrc = Vmat + (size_t)(b * Ss + kbase + lane) * Dd + h * 64;
      unsigned short* vdst = &Vlds[wid][lane * 72];
#pragma unroll
      for (int j = 0; j < 8; ++j) async_copy_b128(vsrc + 8 * j, vdst + 8 * j);
    }

    // scores: two 16x16 tiles over 32 keys, K reduced 64 = 2 WMMAs each
    v8f s0 = {}, s1 = {};
    const unsigned short* k0p = Kmat + (size_t)(b * Ss + kbase + row) * Dd + h * 64;
    const unsigned short* k1p = k0p + (size_t)16 * Dd;
    s0 = __builtin_amdgcn_wmma_f32_16x16x32_bf16(false, qf0, false,
           load_frag16(k0p + half * 16, k0p + half * 16 + 8), (short)0, s0, false, false);
    s0 = __builtin_amdgcn_wmma_f32_16x16x32_bf16(false, qf1, false,
           load_frag16(k0p + 32 + half * 16, k0p + 40 + half * 16), (short)0, s0, false, false);
    s1 = __builtin_amdgcn_wmma_f32_16x16x32_bf16(false, qf0, false,
           load_frag16(k1p + half * 16, k1p + half * 16 + 8), (short)0, s1, false, false);
    s1 = __builtin_amdgcn_wmma_f32_16x16x32_bf16(false, qf1, false,
           load_frag16(k1p + 32 + half * 16, k1p + 40 + half * 16), (short)0, s1, false, false);

    const int c0 = kbase + row, c1 = c0 + 16;
    const float am0 = amask[b * Ss + c0];
    const float am1 = amask[b * Ss + c1];

#pragma unroll
    for (int r = 0; r < 8; ++r) {
      const int gq = qbase + r + 8 * half;
      float v0 = s0[r] * scale;
      float v1 = s1[r] * scale;
      if (c0 > gq || am0 == 0.0f) v0 = -3.0e38f;
      if (c1 > gq || am1 == 0.0f) v1 = -3.0e38f;
      s0[r] = v0; s1[r] = v1;
    }

    // online softmax: row reductions across the 16 lanes of each half-wave
#pragma unroll
    for (int r = 0; r < 8; ++r) {
      float mx = fmaxf(s0[r], s1[r]);
#pragma unroll
      for (int off = 8; off >= 1; off >>= 1) mx = fmaxf(mx, __shfl_xor(mx, off, 16));
      const float mn = fmaxf(m[r], mx);
      const float alpha = __expf(m[r] - mn);
      m[r] = mn;
      const float p0 = __expf(s0[r] - mn);
      const float p1 = __expf(s1[r] - mn);
      s0[r] = p0; s1[r] = p1;
      float rs = p0 + p1;
#pragma unroll
      for (int off = 8; off >= 1; off >>= 1) rs += __shfl_xor(rs, off, 16);
      lsum[r] = lsum[r] * alpha + rs;
#pragma unroll
      for (int cc = 0; cc < 4; ++cc) o[cc][r] *= alpha;
    }

    // P (C layout) -> LDS bf16 -> reload as A fragment (16x32)
    unsigned short* P = Plds[wid];
#pragma unroll
    for (int r = 0; r < 8; ++r) {
      P[(r + 8 * half) * 40 + row]      = f2bf(s0[r]);
      P[(r + 8 * half) * 40 + 16 + row] = f2bf(s1[r]);
    }
    wait_ds0();       // P writes visible (DS in-order within wave)
    wait_async0();    // V tile resident before V^T fragment gathers

    const unsigned short* pb = &P[row * 40];
    const v16bf pf = load_frag16(pb + half * 16, pb + half * 16 + 8);

    // O += P @ V : 4 dim-chunks of 16, V^T fragments gathered from LDS
#pragma unroll
    for (int cc = 0; cc < 4; ++cc) {
      const unsigned short* vb = &Vlds[wid][(half * 16) * 72 + cc * 16 + row];
      const v16bf vf = load_frag_strided(vb, 72);
      o[cc] = __builtin_amdgcn_wmma_f32_16x16x32_bf16(false, pf, false, vf, (short)0, o[cc], false, false);
    }
  }

  // normalize and store bf16 into (B,S,D) attention output
#pragma unroll
  for (int r = 0; r < 8; ++r) {
    const float inv = 1.0f / lsum[r];
    const int gq = qbase + r + 8 * half;
    unsigned short* op = O + (size_t)(b * Ss + gq) * Dd + h * 64 + row;
    op[0]  = f2bf(o[0][r] * inv);
    op[16] = f2bf(o[1][r] * inv);
    op[32] = f2bf(o[2][r] * inv);
    op[48] = f2bf(o[3][r] * inv);
  }
}

// ---------------- orchestration ---------------------------------------------
extern "C" void kernel_launch(void* const* d_in, const int* in_sizes, int n_in,
                              void* d_out, int out_size, void* d_ws, size_t ws_size,
                              hipStream_t stream) {
  (void)in_sizes; (void)n_in; (void)out_size; (void)ws_size;
  const int*   input_ids = (const int*)d_in[0];
  const float* amask     = (const float*)d_in[1];
  const float* tok_emb   = (const float*)d_in[2];
  const float* pos_emb   = (const float*)d_in[3];
  const float* ln1_g = (const float*)d_in[4];
  const float* ln1_b = (const float*)d_in[5];
  const float* Wq = (const float*)d_in[6];   const float* bq = (const float*)d_in[7];
  const float* Wk = (const float*)d_in[8];   const float* bk = (const float*)d_in[9];
  const float* Wv = (const float*)d_in[10];  const float* bv = (const float*)d_in[11];
  const float* Wp = (const float*)d_in[12];  const float* bp = (const float*)d_in[13];
  const float* ln2_g = (const float*)d_in[14];
  const float* ln2_b = (const float*)d_in[15];
  const float* W1 = (const float*)d_in[16];  const float* b1 = (const float*)d_in[17];
  const float* W2 = (const float*)d_in[18];  const float* b2 = (const float*)d_in[19];
  const float* lnf_g = (const float*)d_in[20];
  const float* lnf_b = (const float*)d_in[21];

  // workspace carve-up (~35 MB total)
  char* ws = (char*)d_ws;
  size_t off = 0;
  auto carve = [&](size_t bytes) {
    void* p = ws + off;
    off = (off + bytes + 255) & ~(size_t)255;
    return p;
  };
  int* pos_ids = (int*)carve((size_t)Mrows * 4);
  float* x     = (float*)carve((size_t)Mrows * Dd * 4);
  unsigned short* hbf    = (unsigned short*)carve((size_t)Mrows * Dd * 2);
  unsigned short* qbf    = (unsigned short*)carve((size_t)Mrows * Dd * 2);
  unsigned short* kbf    = (unsigned short*)carve((size_t)Mrows * Dd * 2);
  unsigned short* vbf    = (unsigned short*)carve((size_t)Mrows * Dd * 2);
  unsigned short* attnbf = (unsigned short*)carve((size_t)Mrows * Dd * 2);
  unsigned short* ffbf   = (unsigned short*)carve((size_t)Mrows * DFF * 2);

  auto gemm = [&](const unsigned short* A, const float* W, const float* bias,
                  const float* res, float* oF, unsigned short* oB,
                  int Mm, int Nn, int Kk, int wlayout, int act) {
    dim3 g(Mm / BM, (Nn + BN - 1) / BN);   // M fast => W-tile reuse in L2
    gemm_bf16_kernel<<<g, 256, 0, stream>>>(A, W, bias, res, oF, oB,
                                            Mm, Nn, Kk, wlayout, act);
  };

  pos_scan_kernel<<<Bb, Ss, 0, stream>>>(amask, pos_ids);
  embed_kernel<<<(Mrows * (Dd / 4)) / 256, 256, 0, stream>>>(
      input_ids, pos_ids, tok_emb, pos_emb, x);

  for (int l = 0; l < Ll; ++l) {
    const float* wq = Wq + (size_t)l * Dd * Dd;
    const float* wk = Wk + (size_t)l * Dd * Dd;
    const float* wv = Wv + (size_t)l * Dd * Dd;
    const float* wp = Wp + (size_t)l * Dd * Dd;
    const float* w1 = W1 + (size_t)l * Dd * DFF;
    const float* w2 = W2 + (size_t)l * DFF * Dd;

    ln_kernel<<<Mrows, 256, 0, stream>>>(x, ln1_g + l * Dd, ln1_b + l * Dd, hbf);
    gemm(hbf, wq, bq + l * Dd, nullptr, nullptr, qbf, Mrows, Dd, Dd, 0, 0);
    gemm(hbf, wk, bk + l * Dd, nullptr, nullptr, kbf, Mrows, Dd, Dd, 0, 0);
    gemm(hbf, wv, bv + l * Dd, nullptr, nullptr, vbf, Mrows, Dd, Dd, 0, 0);
    flash_attn_kernel<<<(Bb * Hh * (Ss / 16)) / 4, 128, 0, stream>>>(
        qbf, kbf, vbf, amask, attnbf);
    gemm(attnbf, wp, bp + l * Dd, x, x, nullptr, Mrows, Dd, Dd, 0, 0);
    ln_kernel<<<Mrows, 256, 0, stream>>>(x, ln2_g + l * Dd, ln2_b + l * Dd, hbf);
    gemm(hbf, w1, b1 + l * DFF, nullptr, nullptr, ffbf, Mrows, DFF, Dd, 0, 1);
    gemm(ffbf, w2, b2 + l * Dd, x, x, nullptr, Mrows, Dd, DFF, 0, 0);
  }

  ln_kernel<<<Mrows, 256, 0, stream>>>(x, lnf_g, lnf_b, hbf);
  // tied LM head: logits = lnf(x) @ tok_emb^T  (W stored (N,K) = (V,D))
  gemm(hbf, tok_emb, nullptr, nullptr, (float*)d_out, nullptr,
       Mrows, Vv, Dd, 1, 0);
}
